// TransformerDo_6047313952861
// MI455X (gfx1250) — matrix operand order repaired and checked by reading, not compiled, not run on hardware
//
#include <hip/hip_runtime.h>
#include <hip/hip_bf16.h>
#include <math.h>
#include <stdint.h>

// ---------------------------------------------------------------------------
// Types for CDNA5 WMMA (wave32): v_wmma_f32_16x16x32_bf16
// ---------------------------------------------------------------------------
typedef __bf16 bf16;
typedef __bf16 v16bf __attribute__((ext_vector_type(16)));
typedef __bf16 v8bf  __attribute__((ext_vector_type(8)));
typedef float  v8f   __attribute__((ext_vector_type(8)));

static __device__ __forceinline__ v8f wmma_bf16(v16bf a, v16bf b, v8f c) {
  return __builtin_amdgcn_wmma_f32_16x16x32_bf16(
      /*neg_a=*/false, a, /*neg_b=*/false, b,
      /*c_mod=*/(short)0, c, /*reuse_a=*/false, /*reuse_b=*/false);
}

static __device__ __forceinline__ v16bf cat8(v8bf lo, v8bf hi) {
  return __builtin_shufflevector(lo, hi, 0, 1, 2, 3, 4, 5, 6, 7,
                                 8, 9, 10, 11, 12, 13, 14, 15);
}

// Async global->LDS 16-byte copy (ASYNCcnt-tracked). Generic LDS pointer:
// low 32 bits == LDS byte offset (ISA 10.2 aperture map).
static __device__ __forceinline__ void async_copy16(void* lds, const void* gaddr) {
  unsigned l = (unsigned)(uintptr_t)lds;
  unsigned long long g = (unsigned long long)(uintptr_t)gaddr;
  asm volatile("global_load_async_to_lds_b128 %0, %1, off" ::"v"(l), "v"(g)
               : "memory");
}
// Partial waits: async loads complete in order, so waiting ASYNCcnt<=4 after
// issuing the next tile's 4 copies guarantees the current tile has landed.
static __device__ __forceinline__ void async_wait4() {
  asm volatile("s_wait_asynccnt 4" ::: "memory");
}
static __device__ __forceinline__ void async_wait0() {
  asm volatile("s_wait_asynccnt 0" ::: "memory");
}

#define NEGF (-3.0e38f)

// ---------------------------------------------------------------------------
// Embedding gather: x[b,l,:] = embed[y[b,l],:] + pos[l,:]   (f32 residual)
// ---------------------------------------------------------------------------
__global__ __launch_bounds__(256) void embed_kernel(
    const int* __restrict__ y, const float* __restrict__ embed,
    const float* __restrict__ pos, float* __restrict__ x, int L, int D) {
  int row = blockIdx.x;  // b*L + l
  int tok = y[row];
  int l = row % L;
  const float* e = embed + (size_t)tok * D;
  const float* p = pos + (size_t)l * D;
  float* xo = x + (size_t)row * D;
  for (int i = threadIdx.x; i < D; i += 256) xo[i] = e[i] + p[i];
}

// ---------------------------------------------------------------------------
// LayerNorm over D, one block per row; f32 in, bf16 out (GEMM A operand)
// ---------------------------------------------------------------------------
__global__ __launch_bounds__(256) void ln_kernel(
    const float* __restrict__ x, const float* __restrict__ g,
    bf16* __restrict__ out, int D) {
  int row = blockIdx.x;
  const float* xr = x + (size_t)row * D;
  bf16* orow = out + (size_t)row * D;
  __shared__ float red[256];
  int tid = threadIdx.x;

  float s = 0.f;
  for (int i = tid; i < D; i += 256) s += xr[i];
  red[tid] = s; __syncthreads();
  for (int off = 128; off > 0; off >>= 1) {
    if (tid < off) red[tid] += red[tid + off];
    __syncthreads();
  }
  float mu = red[0] / (float)D;
  __syncthreads();

  float v = 0.f;
  for (int i = tid; i < D; i += 256) { float d = xr[i] - mu; v += d * d; }
  red[tid] = v; __syncthreads();
  for (int off = 128; off > 0; off >>= 1) {
    if (tid < off) red[tid] += red[tid + off];
    __syncthreads();
  }
  float rstd = rsqrtf(red[0] / (float)D + 1e-6f);

  for (int i = tid; i < D; i += 256) orow[i] = (bf16)((xr[i] - mu) * rstd * g[i]);
}

// ---------------------------------------------------------------------------
// Weight transpose + f32->bf16: W[K,N] row-major  ->  Wt[N,K] bf16
// ---------------------------------------------------------------------------
__global__ __launch_bounds__(256) void tcvt_kernel(
    const float* __restrict__ W, bf16* __restrict__ Wt, int K, int N) {
  __shared__ float t[32][33];
  int n0 = blockIdx.x * 32, k0 = blockIdx.y * 32;
  int tx = threadIdx.x & 31, ty = threadIdx.x >> 5;
  #pragma unroll
  for (int r = ty; r < 32; r += 8) {
    int gk = k0 + r, gn = n0 + tx;
    t[r][tx] = (gk < K && gn < N) ? W[(size_t)gk * N + gn] : 0.f;
  }
  __syncthreads();
  #pragma unroll
  for (int r = ty; r < 32; r += 8) {
    int gn = n0 + r, gk = k0 + tx;
    if (gn < N && gk < K) Wt[(size_t)gn * K + gk] = (bf16)t[tx][r];
  }
}

// Plain f32 -> bf16 convert (tied embedding as logits B-matrix [V,K])
__global__ __launch_bounds__(256) void cvt_kernel(
    const float* __restrict__ src, bf16* __restrict__ dst, size_t n) {
  size_t i0 = ((size_t)blockIdx.x * 256 + threadIdx.x) * 4;
  #pragma unroll
  for (int j = 0; j < 4; ++j) {
    size_t i = i0 + j;
    if (i < n) dst[i] = (bf16)src[i];
  }
}

// ---------------------------------------------------------------------------
// GEMM: C = epi(alpha * A @ Bt^T [+ residual])
//   A: bf16 [M,K].  Bt: bf16 [N,K] (pre-transposed).  Block 128x128, K-step
//   32, 8 waves (4m x 2n), wave = 32x64 = 8 wmma/K-step.  Double-buffered
//   async staging overlaps the next tile's DMA with the current tile's WMMA.
//   EPI: 0 f32-out | 1 gelu->bf16 | 2 +res->f32 | 3 bf16-out | 4 bf16 V^T out
// ---------------------------------------------------------------------------
constexpr int TM = 128, TN = 128, TK = 32;

template <int EPI>
__global__ __launch_bounds__(256) void gemm_kernel(
    const bf16* __restrict__ A, const bf16* __restrict__ Bt,
    const float* __restrict__ Res, void* __restrict__ Cv,
    int M, int N, int K, float alpha) {
  __shared__ __align__(64) bf16 As[2][TM][TK];  // 16 KB
  __shared__ __align__(64) bf16 Bs[2][TN][TK];  // 16 KB

  int m0 = blockIdx.y * TM, n0 = blockIdx.x * TN;
  int tid = threadIdx.x;
  int lane = tid & 31, wave = tid >> 5;
  int halfl = lane >> 4, l16 = lane & 15;
  int wm = (wave & 3) * 32, wn = (wave >> 2) * 64;

  v8f acc[2][4] = {};

  auto stage = [&](int k0, int buf) {  // 4 async instructions per wave
    #pragma unroll
    for (int c = 0; c < 2; ++c) {
      int i = tid + c * 256;
      int m = i >> 2, kc = (i & 3) * 8;
      int gm = m0 + m; if (gm >= M) gm = M - 1;   // clamp: junk rows discarded
      async_copy16(&As[buf][m][kc], A + (size_t)gm * K + k0 + kc);
    }
    #pragma unroll
    for (int c = 0; c < 2; ++c) {
      int i = tid + c * 256;
      int n = i >> 2, kc = (i & 3) * 8;
      int gn = n0 + n; if (gn >= N) gn = N - 1;   // clamp: junk cols discarded
      async_copy16(&Bs[buf][n][kc], Bt + (size_t)gn * K + k0 + kc);
    }
  };

  int ntiles = K / TK;  // K is a multiple of 32 for all launches
  stage(0, 0);
  for (int it = 0; it < ntiles; ++it) {
    int cur = it & 1;
    __syncthreads();  // WAR: all waves done reading buf cur^1 (tile it-1)
    if (it + 1 < ntiles) { stage((it + 1) * TK, cur ^ 1); async_wait4(); }
    else async_wait0();
    __syncthreads();  // tile it visible to all waves

    v16bf a[2], b[4];
    #pragma unroll
    for (int f = 0; f < 2; ++f) {
      v8bf lo = *(const v8bf*)&As[cur][wm + f * 16 + l16][halfl * 16];
      v8bf hi = *(const v8bf*)&As[cur][wm + f * 16 + l16][halfl * 16 + 8];
      a[f] = cat8(lo, hi);
    }
    #pragma unroll
    for (int t = 0; t < 4; ++t)
      b[t] = *(const v16bf*)&Bs[cur][wn + t * 16 + l16][halfl * 16];

    #pragma unroll
    for (int f = 0; f < 2; ++f)
      #pragma unroll
      for (int t = 0; t < 4; ++t)
        acc[f][t] = wmma_bf16(a[f], b[t], acc[f][t]);
  }

  // --- epilogue. C layout: row = r + 8*(lane>=16), col = lane&15 ---
  #pragma unroll
  for (int f = 0; f < 2; ++f) {
    #pragma unroll
    for (int t = 0; t < 4; ++t) {
      #pragma unroll
      for (int r = 0; r < 8; ++r) {
        int row = m0 + wm + f * 16 + r + 8 * halfl;
        int col = n0 + wn + t * 16 + l16;
        if (row < M && col < N) {
          float val = acc[f][t][r] * alpha;
          if (EPI == 0) {
            ((float*)Cv)[(size_t)row * N + col] = val;
          } else if (EPI == 1) {
            float x3 = val * val * val;
            val = 0.5f * val *
                  (1.f + tanhf(0.7978845608028654f * (val + 0.044715f * x3)));
            ((bf16*)Cv)[(size_t)row * N + col] = (bf16)val;
          } else if (EPI == 2) {
            ((float*)Cv)[(size_t)row * N + col] = val + Res[(size_t)row * N + col];
          } else if (EPI == 3) {
            ((bf16*)Cv)[(size_t)row * N + col] = (bf16)val;
          } else {  // EPI == 4 : V^T layout [B,H,Dh=64,L=2048]
            int bb = row >> 11, l = row & 2047;
            int h = col >> 6, d = col & 63;
            ((bf16*)Cv)[(((size_t)bb * 12 + h) * 64 + d) * 2048 + l] = (bf16)val;
          }
        }
      }
    }
  }
}

// ---------------------------------------------------------------------------
// Flash attention (causal). q,k: bf16 [B,L,H*64]; vt: bf16 [B,H,64,L];
// o: bf16 [B,L,H*64]. q pre-scaled by 1/8. 4 waves, 16 q-rows/wave.
// Double-buffered async K/V staging; softmax done in A-fragment layout
// (lane = its own row) so only 2 xor-16 merges + 8 broadcasts per tile.
// ---------------------------------------------------------------------------
__global__ __launch_bounds__(128) void flash_kernel(
    const bf16* __restrict__ q, const bf16* __restrict__ k,
    const bf16* __restrict__ vt, bf16* __restrict__ o, int L, int H) {
  constexpr int Dh = 64;
  __shared__ __align__(64) bf16 Ks[2][32][64];   // 2 x 4 KB
  __shared__ __align__(64) bf16 Vs[2][64][32];   // 2 x 4 KB (d-major V^T)
  __shared__ __align__(16) float Sl[4][16][32];  // per-wave f32 S tile, 8 KB

  int tid = threadIdx.x;
  int lane = tid & 31, wave = tid >> 5;
  int halfl = lane >> 4, l16 = lane & 15;
  int b = blockIdx.z, h = blockIdx.y;
  int qblock = blockIdx.x * 64;
  int qbase = qblock + wave * 16;

  size_t sL = (size_t)H * Dh;
  const bf16* qb = q + ((size_t)b * L) * sL + (size_t)h * Dh;
  const bf16* kb = k + ((size_t)b * L) * sL + (size_t)h * Dh;
  const bf16* vtb = vt + ((size_t)b * H + h) * (size_t)Dh * L;

  // Q fragments over d = [0,32) and [32,64)
  v16bf aq[2];
  int qrow = qbase + l16;
  #pragma unroll
  for (int c = 0; c < 2; ++c) {
    v8bf lo = *(const v8bf*)(qb + (size_t)qrow * sL + c * 32 + halfl * 16);
    v8bf hi = *(const v8bf*)(qb + (size_t)qrow * sL + c * 32 + halfl * 16 + 8);
    aq[c] = cat8(lo, hi);
  }

  v8f oacc[4] = {};
  float mi = NEGF, li = 0.f;  // owner-lane stats for row l16 of this wave

  auto stage = [&](int kt, int buf) {  // 4 async instructions per wave
    #pragma unroll
    for (int c = 0; c < 2; ++c) {
      int i = tid + c * 128;
      int key = i >> 3, dc = (i & 7) * 8;
      async_copy16(&Ks[buf][key][dc], kb + (size_t)(kt + key) * sL + dc);
      int d = i >> 2, kc = (i & 3) * 8;
      async_copy16(&Vs[buf][d][kc], vtb + (size_t)d * L + kt + kc);
    }
  };

  int kend = qblock + 63;            // uniform loop bound across the block
  int ntiles = (kend + 32) / 32;     // last staged key = qblock+63 <= L-1
  stage(0, 0);
  for (int it = 0; it < ntiles; ++it) {
    int kt = it * 32;
    int cur = it & 1;
    __syncthreads();  // WAR on buf cur^1
    if (it + 1 < ntiles) { stage(kt + 32, cur ^ 1); async_wait4(); }
    else async_wait0();
    __syncthreads();

    // ---- S = Q K^T (two 16-key halves) ----
    v8f s[2] = {};
    #pragma unroll
    for (int th = 0; th < 2; ++th)
      #pragma unroll
      for (int c = 0; c < 2; ++c) {
        v16bf bk = *(const v16bf*)&Ks[cur][th * 16 + l16][c * 32 + halfl * 16];
        s[th] = wmma_bf16(aq[c], bk, s[th]);
      }

    // ---- spill S (C-layout) to LDS f32 ----
    #pragma unroll
    for (int th = 0; th < 2; ++th)
      #pragma unroll
      for (int r = 0; r < 8; ++r)
        Sl[wave][r + 8 * halfl][th * 16 + l16] = s[th][r];
    asm volatile("s_wait_dscnt 0" ::: "memory");  // same-wave LDS RAW

    // ---- reload in A-layout: lane owns row l16, 16 of 32 keys ----
    float p[16], mx = NEGF;
    int row = qbase + l16;
    #pragma unroll
    for (int j = 0; j < 16; ++j) {
      int kw = halfl * 16 + j;
      float sv = Sl[wave][l16][kw];
      if (kt + kw > row) sv = NEGF;  // causal mask
      p[j] = sv;
      mx = fmaxf(mx, sv);
    }
    mx = fmaxf(mx, __shfl_xor(mx, 16, 32));  // merge the two K-halves
    float nm = fmaxf(mi, mx);
    float alown = __expf(mi - nm);
    float rs = 0.f;
    v16bf pa;  // P already in A-fragment order
    #pragma unroll
    for (int j = 0; j < 16; ++j) {
      float pj = __expf(p[j] - nm);
      rs += pj;
      pa[j] = (bf16)pj;
    }
    rs += __shfl_xor(rs, 16, 32);
    li = li * alown + rs;
    mi = nm;

    // ---- rescale O: broadcast alpha of row (8*halfl + r) from owner lane ----
    #pragma unroll
    for (int r = 0; r < 8; ++r) {
      float a = __shfl(alown, 8 * halfl + r, 32);
      #pragma unroll
      for (int t = 0; t < 4; ++t) oacc[t][r] *= a;
    }

    // ---- O += P V ----
    #pragma unroll
    for (int t = 0; t < 4; ++t) {
      v16bf bv = *(const v16bf*)&Vs[cur][t * 16 + l16][halfl * 16];
      oacc[t] = wmma_bf16(pa, bv, oacc[t]);
    }
  }

  // ---- normalize (broadcast 1/li from owner lanes), store bf16 ----
  float inv = 1.f / li;
  #pragma unroll
  for (int r = 0; r < 8; ++r) {
    float invr = __shfl(inv, 8 * halfl + r, 32);
    #pragma unroll
    for (int t = 0; t < 4; ++t) {
      int row = qbase + r + 8 * halfl;
      int dcol = t * 16 + l16;
      o[((size_t)(b * L + row) * H + h) * Dh + dcol] = (bf16)(oacc[t][r] * invr);
    }
  }
}

// ---------------------------------------------------------------------------
// Host driver
// ---------------------------------------------------------------------------
template <int EPI>
static void launch_gemm(hipStream_t s, const bf16* A, const bf16* Bt,
                        const float* Res, void* C, int M, int N, int K,
                        float alpha) {
  dim3 grid((N + TN - 1) / TN, (M + TM - 1) / TM);
  gemm_kernel<EPI><<<grid, 256, 0, s>>>(A, Bt, Res, C, M, N, K, alpha);
}

extern "C" void kernel_launch(void* const* d_in, const int* in_sizes, int n_in,
                              void* d_out, int out_size, void* d_ws, size_t ws_size,
                              hipStream_t stream) {
  (void)in_sizes; (void)n_in; (void)out_size; (void)ws_size;
  const int*   y     = (const int*)d_in[0];
  const float* embed = (const float*)d_in[1];
  const float* pos   = (const float*)d_in[2];
  const float* ln1   = (const float*)d_in[3];
  const float* wq    = (const float*)d_in[4];
  const float* wk    = (const float*)d_in[5];
  const float* wv    = (const float*)d_in[6];
  const float* wo    = (const float*)d_in[7];
  const float* ln2   = (const float*)d_in[8];
  const float* w1    = (const float*)d_in[9];
  const float* w2    = (const float*)d_in[10];
  const float* lnf   = (const float*)d_in[11];
  float* logits = (float*)d_out;

  constexpr int Bsz = 2, L = 2048, D = 768, H = 12, NL = 4, V = 50257, F = 3072;
  constexpr int BL = Bsz * L;

  char* w = (char*)d_ws;
  float* x   = (float*)w;  w += (size_t)BL * D * 4;
  bf16* hbf  = (bf16*)w;   w += (size_t)BL * D * 2;
  bf16* qbf  = (bf16*)w;   w += (size_t)BL * D * 2;
  bf16* kbf  = (bf16*)w;   w += (size_t)BL * D * 2;
  bf16* vtbf = (bf16*)w;   w += (size_t)BL * D * 2;
  bf16* obf  = (bf16*)w;   w += (size_t)BL * D * 2;
  bf16* zbf  = (bf16*)w;   w += (size_t)BL * F * 2;
  bf16* wtbf = (bf16*)w;   w += (size_t)D * F * 2;
  bf16* ebf  = (bf16*)w;

  embed_kernel<<<BL, 256, 0, stream>>>(y, embed, pos, x, L, D);

  dim3 tDD(D / 32, D / 32), tDF(F / 32, D / 32), tFD(D / 32, F / 32);

  for (int i = 0; i < NL; ++i) {
    ln_kernel<<<BL, 256, 0, stream>>>(x, ln1 + (size_t)i * D, hbf, D);

    tcvt_kernel<<<tDD, 256, 0, stream>>>(wq + (size_t)i * D * D, wtbf, D, D);
    launch_gemm<3>(stream, hbf, wtbf, nullptr, qbf, BL, D, D, 0.125f);
    tcvt_kernel<<<tDD, 256, 0, stream>>>(wk + (size_t)i * D * D, wtbf, D, D);
    launch_gemm<3>(stream, hbf, wtbf, nullptr, kbf, BL, D, D, 1.0f);
    tcvt_kernel<<<tDD, 256, 0, stream>>>(wv + (size_t)i * D * D, wtbf, D, D);
    launch_gemm<4>(stream, hbf, wtbf, nullptr, vtbf, BL, D, D, 1.0f);

    flash_kernel<<<dim3(L / 64, H, Bsz), 128, 0, stream>>>(qbf, kbf, vtbf, obf, L, H);

    tcvt_kernel<<<tDD, 256, 0, stream>>>(wo + (size_t)i * D * D, wtbf, D, D);
    launch_gemm<2>(stream, obf, wtbf, x, x, BL, D, D, 1.0f);

    ln_kernel<<<BL, 256, 0, stream>>>(x, ln2 + (size_t)i * D, hbf, D);
    tcvt_kernel<<<tDF, 256, 0, stream>>>(w1 + (size_t)i * D * F, wtbf, D, F);
    launch_gemm<1>(stream, hbf, wtbf, nullptr, zbf, BL, F, D, 1.0f);
    tcvt_kernel<<<tFD, 256, 0, stream>>>(w2 + (size_t)i * F * D, wtbf, F, D);
    launch_gemm<2>(stream, zbf, wtbf, x, x, BL, D, F, 1.0f);
  }

  ln_kernel<<<BL, 256, 0, stream>>>(x, lnf, hbf, D);
  cvt_kernel<<<(int)(((size_t)V * D + 1023) / 1024), 256, 0, stream>>>(
      embed, ebf, (size_t)V * D);
  launch_gemm<0>(stream, hbf, ebf, nullptr, logits, BL, V, D, 1.0f);
}
